// MiniGPT_59176059404473
// MI455X (gfx1250) — matrix operand order repaired and checked by reading, not compiled, not run on hardware
//
#include <hip/hip_runtime.h>
#include <hip/hip_bf16.h>
#include <math.h>

// ---------------------------------------------------------------------------
// Model dims (fixed by the reference)
// ---------------------------------------------------------------------------
#define GPT_B 2
#define GPT_S 2048
#define GPT_D 1024
#define GPT_H 16
#define GPT_DK 64
#define GPT_L 6
#define GPT_F 4096
#define GPT_V 32000
#define GPT_M (GPT_B * GPT_S)   // 4096 rows

typedef __attribute__((ext_vector_type(16))) __bf16 v16bf;
typedef __attribute__((ext_vector_type(8)))  float  v8f;

#define WMMA_BF16(a, b, c) \
  __builtin_amdgcn_wmma_f32_16x16x32_bf16(false, (a), false, (b), (short)0, (c), false, false)

union FragU { unsigned int u[8]; v16bf v; };

__device__ __forceinline__ unsigned short f32_to_bf16(float x) {
  union { __bf16 h; unsigned short u; } cv;
  cv.h = (__bf16)x;
  return cv.u;
}

__device__ __forceinline__ unsigned lds_offset_of(const void* p) {
  // Generic pointers to LDS carry the wave-relative offset in the low 32 bits.
  return (unsigned)(uintptr_t)p;
}

// Async DMA: copy 8 bytes from global to LDS (tracked by ASYNCcnt, no VGPRs).
__device__ __forceinline__ void async_ld_b64(unsigned lds_byte_off, const void* gaddr) {
  asm volatile("global_load_async_to_lds_b64 %0, %1, off"
               :: "v"(lds_byte_off), "v"((unsigned long long)(uintptr_t)gaddr)
               : "memory");
}

__device__ __forceinline__ void wait_asynccnt0() {
  asm volatile("s_wait_asynccnt 0x0" ::: "memory");
}

// A-fragment (16x32 bf16, MxK) from LDS row-major [m][k] (k contiguous).
__device__ __forceinline__ v16bf load_frag_a(const unsigned short* base, int stride, int lane) {
  const int m = lane & 15, half = lane >> 4;
  const unsigned short* row = base + m * stride;
  FragU f;
#pragma unroll
  for (int j = 0; j < 4; ++j)
    f.u[j] = *(const unsigned int*)(row + half * 8 + 2 * j);
#pragma unroll
  for (int j = 0; j < 4; ++j)
    f.u[4 + j] = *(const unsigned int*)(row + 16 + half * 8 + 2 * j);
  return f.v;
}

// B-fragment (32x16 bf16, KxN) from LDS stored TRANSPOSED [n][k] (k contiguous).
__device__ __forceinline__ v16bf load_frag_b(const unsigned short* base, int stride, int lane) {
  const int n = lane & 15, half = lane >> 4;
  const unsigned short* row = base + n * stride;
  FragU f;
#pragma unroll
  for (int j = 0; j < 4; ++j)
    f.u[j] = *(const unsigned int*)(row + half * 8 + 2 * j);
#pragma unroll
  for (int j = 0; j < 4; ++j)
    f.u[4 + j] = *(const unsigned int*)(row + 16 + half * 8 + 2 * j);
  return f.v;
}

// ---------------------------------------------------------------------------
// Embedding
// ---------------------------------------------------------------------------
__global__ __launch_bounds__(256) void gpt_embed(const int* __restrict__ tokens,
                                                 const float* __restrict__ tok_emb,
                                                 const float* __restrict__ pos_emb,
                                                 float* __restrict__ x) {
  const int bs = blockIdx.x;
  const int s  = bs & (GPT_S - 1);
  const int tok = tokens[bs];
  const float* te = tok_emb + (size_t)tok * GPT_D;
  const float* pe = pos_emb + (size_t)s * GPT_D;
  float* xr = x + (size_t)bs * GPT_D;
  for (int d = threadIdx.x; d < GPT_D; d += 256) xr[d] = te[d] + pe[d];
}

// ---------------------------------------------------------------------------
// LayerNorm (fp32 in) -> bf16 out. One block per row, D = 1024.
// ---------------------------------------------------------------------------
__global__ __launch_bounds__(256) void gpt_layernorm(const float* __restrict__ x,
                                                     const float* __restrict__ g,
                                                     const float* __restrict__ bta,
                                                     unsigned short* __restrict__ out) {
  const int row = blockIdx.x;
  const float* xr = x + (size_t)row * GPT_D;
  __shared__ float s_sum[8], s_sq[8];
  float sum = 0.f, sq = 0.f;
  for (int d = threadIdx.x; d < GPT_D; d += 256) {
    float v = xr[d];
    sum += v; sq += v * v;
  }
#pragma unroll
  for (int off = 16; off > 0; off >>= 1) {
    sum += __shfl_xor(sum, off);
    sq  += __shfl_xor(sq, off);
  }
  const int wave = threadIdx.x >> 5;
  if ((threadIdx.x & 31) == 0) { s_sum[wave] = sum; s_sq[wave] = sq; }
  __syncthreads();
  float ts = 0.f, tq = 0.f;
#pragma unroll
  for (int i = 0; i < 8; ++i) { ts += s_sum[i]; tq += s_sq[i]; }
  const float mean = ts * (1.f / GPT_D);
  const float var  = tq * (1.f / GPT_D) - mean * mean;
  const float rstd = rsqrtf(var + 1e-5f);
  unsigned short* orow = out + (size_t)row * GPT_D;
  for (int d = threadIdx.x; d < GPT_D; d += 256)
    orow[d] = f32_to_bf16((xr[d] - mean) * rstd * g[d] + bta[d]);
}

// ---------------------------------------------------------------------------
// Weight pre-pass: W[K,N] fp32 -> Wt[N,K] bf16 (32x32 LDS tile transpose).
// ---------------------------------------------------------------------------
__global__ __launch_bounds__(256) void gpt_convert_wt(const float* __restrict__ W,
                                                      unsigned short* __restrict__ Wt,
                                                      int K, int N) {
  __shared__ float tile[32][33];
  const int n0 = blockIdx.x * 32;
  const int k0 = blockIdx.y * 32;
  const int tx = threadIdx.x & 31;
  const int ty = threadIdx.x >> 5;   // 0..7
#pragma unroll
  for (int r = 0; r < 32; r += 8)
    tile[r + ty][tx] = W[(size_t)(k0 + r + ty) * N + (n0 + tx)];
  __syncthreads();
#pragma unroll
  for (int r = 0; r < 32; r += 8)
    Wt[(size_t)(n0 + r + ty) * K + (k0 + tx)] = f32_to_bf16(tile[tx][r + ty]);
}

// ---------------------------------------------------------------------------
// GEMM: C[M,N] = epilogue( A_bf16[M,K] @ Wt_bf16[N,K]^T )
// Tile 128x128x32, 256 threads = 8 waves (2x4); async LDS DMA, double-buffered.
// ---------------------------------------------------------------------------
#define GST 36   // LDS row stride in ushorts (72 B)

#define MODE_BF16           0   // bf16 out
#define MODE_BF16_BIAS_GELU 1   // bf16 out, +bias, exact gelu
#define MODE_F32_RESID      2   // f32 out, +resid
#define MODE_F32_BIAS_RESID 3   // f32 out, +bias, +resid
#define MODE_F32            4   // f32 out
#define MODE_BF16_SCALE     5   // bf16 out, *0.125 (attention 1/sqrt(DK) folded into Q)

__device__ __forceinline__ void async_copy_tile(const unsigned short* gsrc, int src_stride,
                                                int k0, unsigned lds_base, int tid) {
#pragma unroll
  for (int i = 0; i < 4; ++i) {
    const int c = tid + i * 256;        // 0..1023 : 8-byte chunk id
    const int row = c >> 3;
    const int q   = c & 7;
    const unsigned short* g = gsrc + (size_t)row * src_stride + k0 + q * 4;
    const unsigned lds = lds_base + (unsigned)((row * GST + q * 4) * 2);
    async_ld_b64(lds, g);
  }
}

template <int MODE>
__global__ __launch_bounds__(256) void gpt_gemm_wmma(
    const unsigned short* __restrict__ A,    // [M,K] bf16
    const unsigned short* __restrict__ Wt,   // [N,K] bf16 (pre-transposed)
    const float* __restrict__ bias,          // [N]
    const float* __restrict__ resid,         // [M,N]
    float* __restrict__ outf,                // [M,N]
    unsigned short* __restrict__ outb,       // [M,N] bf16
    int M, int N, int K) {
  __shared__ unsigned short As[2][128 * GST];
  __shared__ unsigned short Bts[2][128 * GST];

  const int tid  = threadIdx.x;
  const int lane = tid & 31;
  const int wave = tid >> 5;
  const int wm = wave >> 2;   // 0..1 -> 64-row strip
  const int wn = wave & 3;    // 0..3 -> 32-col strip
  const int n0 = blockIdx.x * 128;
  const int m0 = blockIdx.y * 128;

  const unsigned short* Ag = A  + (size_t)m0 * K;
  const unsigned short* Bg = Wt + (size_t)n0 * K;
  const unsigned as_base[2] = { lds_offset_of(&As[0][0]),  lds_offset_of(&As[1][0]) };
  const unsigned bs_base[2] = { lds_offset_of(&Bts[0][0]), lds_offset_of(&Bts[1][0]) };

  v8f acc[4][2];
#pragma unroll
  for (int i = 0; i < 4; ++i)
#pragma unroll
    for (int j = 0; j < 2; ++j)
      acc[i][j] = v8f{0.f, 0.f, 0.f, 0.f, 0.f, 0.f, 0.f, 0.f};

  async_copy_tile(Ag, K, 0, as_base[0], tid);
  async_copy_tile(Bg, K, 0, bs_base[0], tid);

  int cur = 0;
  for (int k0 = 0; k0 < K; k0 += 32) {
    wait_asynccnt0();
    __syncthreads();
    if (k0 + 32 < K) {
      async_copy_tile(Ag, K, k0 + 32, as_base[cur ^ 1], tid);
      async_copy_tile(Bg, K, k0 + 32, bs_base[cur ^ 1], tid);
    }

    const unsigned short* Asc = &As[cur][0];
    const unsigned short* Bsc = &Bts[cur][0];
    v16bf af[4], bfr[2];
#pragma unroll
    for (int mf = 0; mf < 4; ++mf)
      af[mf] = load_frag_a(Asc + (wm * 64 + mf * 16) * GST, GST, lane);
#pragma unroll
    for (int nf = 0; nf < 2; ++nf)
      bfr[nf] = load_frag_b(Bsc + (wn * 32 + nf * 16) * GST, GST, lane);
#pragma unroll
    for (int mf = 0; mf < 4; ++mf)
#pragma unroll
      for (int nf = 0; nf < 2; ++nf)
        acc[mf][nf] = WMMA_BF16(af[mf], bfr[nf], acc[mf][nf]);

    __syncthreads();
    cur ^= 1;
  }

  const int half = lane >> 4, nl = lane & 15;
#pragma unroll
  for (int mf = 0; mf < 4; ++mf) {
#pragma unroll
    for (int nf = 0; nf < 2; ++nf) {
      const int n = n0 + wn * 32 + nf * 16 + nl;
      float bv = 0.f;
      if (MODE == MODE_BF16_BIAS_GELU || MODE == MODE_F32_BIAS_RESID) bv = bias[n];
#pragma unroll
      for (int r = 0; r < 8; ++r) {
        const int m = m0 + wm * 64 + mf * 16 + half * 8 + r;
        const size_t idx = (size_t)m * N + n;
        float v = acc[mf][nf][r] + bv;
        if (MODE == MODE_BF16_SCALE) v *= 0.125f;   // 1/sqrt(DK)
        if (MODE == MODE_BF16_BIAS_GELU)
          v = 0.5f * v * (1.f + erff(v * 0.70710678118654752f));
        if (MODE == MODE_F32_RESID || MODE == MODE_F32_BIAS_RESID) v += resid[idx];
        if (MODE == MODE_BF16 || MODE == MODE_BF16_BIAS_GELU || MODE == MODE_BF16_SCALE)
          outb[idx] = f32_to_bf16(v);
        else
          outf[idx] = v;
      }
    }
  }
}

// ---------------------------------------------------------------------------
// Flash attention with WMMA. Grid (S/64, B*H), 128 threads = 4 waves.
// Q pre-scaled by 1/sqrt(DK); mask applied only on the diagonal key tile.
// Q/K tiles staged via async LDS DMA; V transposed in-flight (overlaps K DMA).
// ---------------------------------------------------------------------------
#define AST 68   // LDS row stride in ushorts (136 B, 8B-aligned, conflict-free rows)

// Straight 64x64 bf16 tile copy via async DMA (row stride = GPT_D elems).
__device__ __forceinline__ void async_copy_att_tile(const unsigned short* gsrc,
                                                    unsigned lds_base, int tid) {
#pragma unroll
  for (int i = 0; i < 8; ++i) {
    const int c = tid + i * 128;        // 0..1023 : 8-byte chunk id
    const int row = c >> 4;
    const int q   = c & 15;
    const unsigned short* g = gsrc + (size_t)row * GPT_D + q * 4;
    const unsigned lds = lds_base + (unsigned)((row * AST + q * 4) * 2);
    async_ld_b64(lds, g);
  }
}

__global__ __launch_bounds__(128) void gpt_attention_wmma(
    const unsigned short* __restrict__ Q,   // [B,S,D] bf16 (pre-scaled)
    const unsigned short* __restrict__ Kk,  // [B,S,D] bf16
    const unsigned short* __restrict__ Vv,  // [B,S,D] bf16
    unsigned short* __restrict__ O) {       // [B,S,D] bf16
  __shared__ unsigned short Qs[64 * AST];       // [q][d]
  __shared__ unsigned short Ks[64 * AST];       // [k][d]
  __shared__ unsigned short Vts[64 * AST];      // [d][k] (transposed)
  __shared__ unsigned short Ps[4][16 * AST];    // per-wave P scratch [m][k]

  const int tid  = threadIdx.x;
  const int lane = tid & 31;
  const int wave = tid >> 5;
  const int half = lane >> 4, nl = lane & 15;
  const int qt = blockIdx.x;
  const int bh = blockIdx.y;
  const int b = bh / GPT_H, h = bh % GPT_H;
  const int q0 = qt * 64;
  const size_t base = (size_t)b * GPT_S * GPT_D + (size_t)h * GPT_DK;

  const unsigned qs_base = lds_offset_of(&Qs[0]);
  const unsigned ks_base = lds_offset_of(&Ks[0]);

  // Q tile via async DMA.
  async_copy_att_tile(Q + base + (size_t)q0 * GPT_D, qs_base, tid);
  wait_asynccnt0();
  __syncthreads();

  const v16bf qa0 = load_frag_a(&Qs[(wave * 16) * AST], AST, lane);        // d 0..31
  const v16bf qa1 = load_frag_a(&Qs[(wave * 16) * AST + 32], AST, lane);   // d 32..63

  float m_run[8], l_run[8];
  v8f accd[4];
#pragma unroll
  for (int r = 0; r < 8; ++r) { m_run[r] = -INFINITY; l_run[r] = 0.f; }
#pragma unroll
  for (int df = 0; df < 4; ++df)
    accd[df] = v8f{0.f, 0.f, 0.f, 0.f, 0.f, 0.f, 0.f, 0.f};

  for (int kt = 0; kt <= qt; ++kt) {
    const int k0 = kt * 64;
    // K tile: async DMA, overlapped with the manual V transpose below.
    async_copy_att_tile(Kk + base + (size_t)k0 * GPT_D, ks_base, tid);
    // V tile transposed -> Vts[d][k]
#pragma unroll
    for (int it = 0; it < 16; ++it) {
      int i = tid + it * 128;
      int r = i >> 5, c2 = i & 31;        // r = key row, c2 = d pair
      unsigned int val =
          *(const unsigned int*)(Vv + base + (size_t)(k0 + r) * GPT_D + 2 * c2);
      Vts[(2 * c2 + 0) * AST + r] = (unsigned short)(val & 0xffffu);
      Vts[(2 * c2 + 1) * AST + r] = (unsigned short)(val >> 16);
    }
    wait_asynccnt0();
    __syncthreads();

    // Scores (Q already carries the 1/sqrt(DK) scale).
    v8f sc[4];
#pragma unroll
    for (int nf = 0; nf < 4; ++nf) {
      v16bf kb0 = load_frag_b(&Ks[(nf * 16) * AST], AST, lane);        // d 0..31
      v16bf kb1 = load_frag_b(&Ks[(nf * 16) * AST + 32], AST, lane);   // d 32..63
      v8f z = v8f{0.f, 0.f, 0.f, 0.f, 0.f, 0.f, 0.f, 0.f};
      sc[nf] = WMMA_BF16(qa0, kb0, z);
      sc[nf] = WMMA_BF16(qa1, kb1, sc[nf]);
    }

    // Causal mask only possible on the diagonal tile (uniform branch).
    if (kt == qt) {
#pragma unroll
      for (int nf = 0; nf < 4; ++nf) {
        const int kg = k0 + nf * 16 + nl;
#pragma unroll
        for (int r = 0; r < 8; ++r) {
          const int mg = q0 + wave * 16 + half * 8 + r;
          if (kg > mg) sc[nf][r] = -INFINITY;
        }
      }
    }

    float mx[8];
#pragma unroll
    for (int r = 0; r < 8; ++r) mx[r] = -INFINITY;
#pragma unroll
    for (int nf = 0; nf < 4; ++nf)
#pragma unroll
      for (int r = 0; r < 8; ++r)
        mx[r] = fmaxf(mx[r], sc[nf][r]);
#pragma unroll
    for (int r = 0; r < 8; ++r) {
#pragma unroll
      for (int off = 1; off < 16; off <<= 1)
        mx[r] = fmaxf(mx[r], __shfl_xor(mx[r], off));
    }

    float cf[8], rs[8];
#pragma unroll
    for (int r = 0; r < 8; ++r) {
      float nm = fmaxf(m_run[r], mx[r]);
      cf[r] = expf(m_run[r] - nm);
      m_run[r] = nm;
      rs[r] = 0.f;
    }
#pragma unroll
    for (int nf = 0; nf < 4; ++nf)
#pragma unroll
      for (int r = 0; r < 8; ++r) {
        float p = expf(sc[nf][r] - m_run[r]);
        sc[nf][r] = p;
        rs[r] += p;
      }
#pragma unroll
    for (int r = 0; r < 8; ++r) {
#pragma unroll
      for (int off = 1; off < 16; off <<= 1)
        rs[r] += __shfl_xor(rs[r], off);
      l_run[r] = l_run[r] * cf[r] + rs[r];
    }
#pragma unroll
    for (int df = 0; df < 4; ++df)
#pragma unroll
      for (int r = 0; r < 8; ++r)
        accd[df][r] *= cf[r];

#pragma unroll
    for (int nf = 0; nf < 4; ++nf)
#pragma unroll
      for (int r = 0; r < 8; ++r)
        Ps[wave][(half * 8 + r) * AST + nf * 16 + nl] = f32_to_bf16(sc[nf][r]);
    __syncthreads();

    v16bf pa0 = load_frag_a(&Ps[wave][0], AST, lane);       // k 0..31
    v16bf pa1 = load_frag_a(&Ps[wave][32], AST, lane);      // k 32..63
#pragma unroll
    for (int df = 0; df < 4; ++df) {
      v16bf vb0 = load_frag_b(&Vts[(df * 16) * AST], AST, lane);       // k 0..31
      v16bf vb1 = load_frag_b(&Vts[(df * 16) * AST + 32], AST, lane);  // k 32..63
      accd[df] = WMMA_BF16(pa0, vb0, accd[df]);
      accd[df] = WMMA_BF16(pa1, vb1, accd[df]);
    }
    __syncthreads();
  }

#pragma unroll
  for (int df = 0; df < 4; ++df) {
    const int d = df * 16 + nl;
#pragma unroll
    for (int r = 0; r < 8; ++r) {
      const int mg = q0 + wave * 16 + half * 8 + r;
      float ov = accd[df][r] / l_run[r];
      O[base + (size_t)mg * GPT_D + d] = f32_to_bf16(ov);
    }
  }
}

// ---------------------------------------------------------------------------
// Host orchestration
// ---------------------------------------------------------------------------
extern "C" void kernel_launch(void* const* d_in, const int* in_sizes, int n_in,
                              void* d_out, int out_size, void* d_ws, size_t ws_size,
                              hipStream_t stream) {
  (void)in_sizes; (void)n_in; (void)out_size; (void)ws_size;

  const int*   tokens  = (const int*)  d_in[0];
  const float* tok_emb = (const float*)d_in[1];
  const float* pos_emb = (const float*)d_in[2];
  const float* Wq      = (const float*)d_in[3];
  const float* Wk      = (const float*)d_in[4];
  const float* Wv      = (const float*)d_in[5];
  const float* Wo      = (const float*)d_in[6];
  const float* ln1_g   = (const float*)d_in[7];
  const float* ln1_b   = (const float*)d_in[8];
  const float* ln2_g   = (const float*)d_in[9];
  const float* ln2_b   = (const float*)d_in[10];
  const float* W1      = (const float*)d_in[11];
  const float* b1      = (const float*)d_in[12];
  const float* W2      = (const float*)d_in[13];
  const float* b2      = (const float*)d_in[14];
  const float* lnf_g   = (const float*)d_in[15];
  const float* lnf_b   = (const float*)d_in[16];
  const float* Wout    = (const float*)d_in[17];
  float* logits        = (float*)d_out;

  char* ws = (char*)d_ws;
  float*          x   = (float*)ws;                                         // 16 MB
  unsigned short* hb  = (unsigned short*)(ws + (size_t)16 * 1024 * 1024);   // 8 MB
  unsigned short* qb  = (unsigned short*)(ws + (size_t)24 * 1024 * 1024);   // 8 MB
  unsigned short* kb  = (unsigned short*)(ws + (size_t)32 * 1024 * 1024);   // 8 MB
  unsigned short* vb  = (unsigned short*)(ws + (size_t)40 * 1024 * 1024);   // 8 MB
  unsigned short* cb  = (unsigned short*)(ws + (size_t)48 * 1024 * 1024);   // 8 MB
  unsigned short* ffh = (unsigned short*)(ws + (size_t)56 * 1024 * 1024);   // 32 MB
  unsigned short* wt  = (unsigned short*)(ws + (size_t)88 * 1024 * 1024);   // 64 MB

  const dim3 blk256(256), blk128(128);
  const dim3 gemmDD(GPT_D / 128, GPT_M / 128);
  const dim3 gemmDF(GPT_F / 128, GPT_M / 128);
  const dim3 gemmDV(GPT_V / 128, GPT_M / 128);
  const dim3 cvtDD(GPT_D / 32, GPT_D / 32);
  const dim3 cvtDF(GPT_F / 32, GPT_D / 32);
  const dim3 cvtFD(GPT_D / 32, GPT_F / 32);
  const dim3 cvtDV(GPT_V / 32, GPT_D / 32);
  const dim3 attnGrid(GPT_S / 64, GPT_B * GPT_H);

  gpt_embed<<<GPT_M, blk256, 0, stream>>>(tokens, tok_emb, pos_emb, x);

  for (int l = 0; l < GPT_L; ++l) {
    const float* wqp = Wq + (size_t)l * GPT_D * GPT_D;
    const float* wkp = Wk + (size_t)l * GPT_D * GPT_D;
    const float* wvp = Wv + (size_t)l * GPT_D * GPT_D;
    const float* wop = Wo + (size_t)l * GPT_D * GPT_D;
    const float* w1p = W1 + (size_t)l * GPT_D * GPT_F;
    const float* w2p = W2 + (size_t)l * GPT_F * GPT_D;

    gpt_layernorm<<<GPT_M, blk256, 0, stream>>>(x, ln1_g + l * GPT_D, ln1_b + l * GPT_D, hb);

    // Q projection carries the 1/sqrt(DK) scale.
    gpt_convert_wt<<<cvtDD, blk256, 0, stream>>>(wqp, wt, GPT_D, GPT_D);
    gpt_gemm_wmma<MODE_BF16_SCALE><<<gemmDD, blk256, 0, stream>>>(
        hb, wt, nullptr, nullptr, nullptr, qb, GPT_M, GPT_D, GPT_D);
    gpt_convert_wt<<<cvtDD, blk256, 0, stream>>>(wkp, wt, GPT_D, GPT_D);
    gpt_gemm_wmma<MODE_BF16><<<gemmDD, blk256, 0, stream>>>(
        hb, wt, nullptr, nullptr, nullptr, kb, GPT_M, GPT_D, GPT_D);
    gpt_convert_wt<<<cvtDD, blk256, 0, stream>>>(wvp, wt, GPT_D, GPT_D);
    gpt_gemm_wmma<MODE_BF16><<<gemmDD, blk256, 0, stream>>>(
        hb, wt, nullptr, nullptr, nullptr, vb, GPT_M, GPT_D, GPT_D);

    gpt_attention_wmma<<<attnGrid, blk128, 0, stream>>>(qb, kb, vb, cb);

    // x = x + ctx @ Wo
    gpt_convert_wt<<<cvtDD, blk256, 0, stream>>>(wop, wt, GPT_D, GPT_D);
    gpt_gemm_wmma<MODE_F32_RESID><<<gemmDD, blk256, 0, stream>>>(
        cb, wt, nullptr, x, x, nullptr, GPT_M, GPT_D, GPT_D);

    gpt_layernorm<<<GPT_M, blk256, 0, stream>>>(x, ln2_g + l * GPT_D, ln2_b + l * GPT_D, hb);

    // h = gelu(ln2(x) @ W1 + b1)
    gpt_convert_wt<<<cvtDF, blk256, 0, stream>>>(w1p, wt, GPT_D, GPT_F);
    gpt_gemm_wmma<MODE_BF16_BIAS_GELU><<<gemmDF, blk256, 0, stream>>>(
        hb, wt, b1 + l * GPT_F, nullptr, nullptr, ffh, GPT_M, GPT_F, GPT_D);
    // x = x + h @ W2 + b2
    gpt_convert_wt<<<cvtFD, blk256, 0, stream>>>(w2p, wt, GPT_F, GPT_D);
    gpt_gemm_wmma<MODE_F32_BIAS_RESID><<<gemmDD, blk256, 0, stream>>>(
        ffh, wt, b2 + l * GPT_D, x, x, nullptr, GPT_M, GPT_D, GPT_F);
  }

  gpt_layernorm<<<GPT_M, blk256, 0, stream>>>(x, lnf_g, lnf_b, hb);
  gpt_convert_wt<<<cvtDV, blk256, 0, stream>>>(Wout, wt, GPT_D, GPT_V);
  gpt_gemm_wmma<MODE_F32><<<gemmDV, blk256, 0, stream>>>(
      hb, wt, nullptr, nullptr, logits, nullptr, GPT_M, GPT_V, GPT_D);
}
